// MLP_63359357551382
// MI455X (gfx1250) — matrix operand order repaired and checked by reading, not compiled, not run on hardware
//
#include <hip/hip_runtime.h>
#include <hip/hip_bf16.h>

// Problem constants (match reference)
constexpr int   Hd  = 1024;   // hidden
constexpr int   Id  = 1024;   // intermediate
constexpr int   Ed  = 16;     // experts
constexpr int   Kt  = 4;      // experts per token
constexpr int   Td  = 1024;   // tokens (2*512)
constexpr float LIM = 7.0f;
constexpr float AL  = 1.702f;
constexpr float EPSV = 1e-5f;

constexpr int   BPITCH = 40;  // LDS B-tile row pitch in halves (80B: 16B-aligned, conflict-free)

typedef __attribute__((ext_vector_type(16))) __bf16          v16bf;
typedef __attribute__((ext_vector_type(16))) unsigned short  v16u;
typedef __attribute__((ext_vector_type(8)))  float           v8f;

union BF16x16 { v16u u; v16bf b; unsigned int d[8]; };

__device__ __forceinline__ unsigned short f2bf(float f) {
  unsigned int u = __float_as_uint(f);
  unsigned int r = u + 0x7fffu + ((u >> 16) & 1u);   // round-to-nearest-even
  return (unsigned short)(r >> 16);
}

// ---------------------------------------------------------------------------
// Kernel 1: RMSNorm (fp32) -> xn bf16, router logits, top-4 softmax, scatter
// ---------------------------------------------------------------------------
__global__ void rmsnorm_router(const float* __restrict__ x,
                               const float* __restrict__ nscale,
                               const float* __restrict__ wg,
                               const float* __restrict__ bg,
                               unsigned short* __restrict__ xnbf,
                               int* __restrict__ cnt,
                               int* __restrict__ tok_list,
                               int* __restrict__ row_list,
                               float* __restrict__ gate_tk) {
  const int t   = blockIdx.x;
  const int tid = threadIdx.x;                  // 128 threads
  __shared__ float red[128];
  __shared__ float lacc[128 * 16];
  __shared__ float logits[16];

  const float* xr = x + (size_t)t * Hd;
  float v[8];
  float ss = 0.f;
#pragma unroll
  for (int j = 0; j < 8; ++j) { v[j] = xr[tid + j * 128]; ss += v[j] * v[j]; }
  red[tid] = ss;
  __syncthreads();
  for (int s = 64; s > 0; s >>= 1) {
    if (tid < s) red[tid] += red[tid + s];
    __syncthreads();
  }
  const float rinv = rsqrtf(red[0] / (float)Hd + EPSV);

  float acc[16];
#pragma unroll
  for (int e = 0; e < 16; ++e) acc[e] = 0.f;
#pragma unroll
  for (int j = 0; j < 8; ++j) {
    const int i = tid + j * 128;
    const float xn = v[j] * rinv * nscale[i];
    xnbf[(size_t)t * Hd + i] = f2bf(xn);
    const float* wr = wg + (size_t)i * Ed;
#pragma unroll
    for (int e = 0; e < 16; ++e) acc[e] = fmaf(xn, wr[e], acc[e]);
  }
#pragma unroll
  for (int e = 0; e < 16; ++e) lacc[tid * 16 + e] = acc[e];
  __syncthreads();
  if (tid < 16) {                                // deterministic column sums
    float s = bg[tid];
    for (int i = 0; i < 128; ++i) s += lacc[i * 16 + tid];
    logits[tid] = s;
  }
  __syncthreads();
  if (tid == 0) {
    unsigned used = 0;
    int   idx[Kt];
    float val[Kt];
    for (int k = 0; k < Kt; ++k) {
      float best = -3.4e38f; int bi = 0;
      for (int e = 0; e < 16; ++e)
        if (!((used >> e) & 1u) && logits[e] > best) { best = logits[e]; bi = e; }
      used |= 1u << bi; idx[k] = bi; val[k] = best;
    }
    const float m = val[0];
    float g[Kt], sum = 0.f;
    for (int k = 0; k < Kt; ++k) { g[k] = __expf(val[k] - m); sum += g[k]; }
    for (int k = 0; k < Kt; ++k) {
      const float gate = g[k] / sum;
      const int e = idx[k];
      const int pos = atomicAdd(cnt + e, 1);
      tok_list[e * Td + pos] = t;
      row_list[e * Td + pos] = t * Kt + k;
      gate_tk[t * Kt + k] = gate;
    }
  }
}

// ---------------------------------------------------------------------------
// GEMM building blocks: 64x64 macro-tile, 128 threads (4 waves)
//   wave w -> rows w*16..w*16+15, 4 column subtiles of 16
//   double-buffered LDS B tile; per k-step: 4 WMMAs/wave
// ---------------------------------------------------------------------------
struct WTile { float4 w[4]; };   // per-thread slice of a 32(K) x 64(N) f32 tile

__device__ __forceinline__ void load_A_frag(BF16x16& A, const unsigned short* arow) {
  // A 16x32 bf16 layout: per lane 32 contiguous bytes (halves kg*16+0..15)
  const uint4 a0 = *(const uint4*)(arow);
  const uint4 a1 = *(const uint4*)(arow + 8);
  A.d[0] = a0.x; A.d[1] = a0.y; A.d[2] = a0.z; A.d[3] = a0.w;
  A.d[4] = a1.x; A.d[5] = a1.y; A.d[6] = a1.z; A.d[7] = a1.w;
}

__device__ __forceinline__ void load_W(WTile& t, const float* wbase, int k0,
                                       int nstride, int tid) {
#pragma unroll
  for (int g = 0; g < 2; ++g) {
    const int p = g * 128 + tid;               // 0..255: (k-pair, n-quad)
    const int k = (p >> 4) * 2;
    const int n = (p & 15) * 4;
    t.w[2 * g + 0] = *(const float4*)(wbase + (size_t)(k0 + k)     * nstride + n);
    t.w[2 * g + 1] = *(const float4*)(wbase + (size_t)(k0 + k + 1) * nstride + n);
  }
}

__device__ __forceinline__ void store_W(const WTile& t, unsigned short* Blds, int tid) {
#pragma unroll
  for (int g = 0; g < 2; ++g) {
    const int p = g * 128 + tid;
    const int k = (p >> 4) * 2;
    const int n = (p & 15) * 4;
    const float* lo = (const float*)&t.w[2 * g + 0];
    const float* hi = (const float*)&t.w[2 * g + 1];
#pragma unroll
    for (int j = 0; j < 4; ++j) {              // pack (k, k+1) halves -> b32 store
      const unsigned int pk = (unsigned int)f2bf(lo[j]) |
                              ((unsigned int)f2bf(hi[j]) << 16);
      *(unsigned int*)(Blds + (n + j) * BPITCH + k) = pk;
    }
  }
}

__device__ __forceinline__ void load_B_frag(BF16x16& B, const unsigned short* Blds,
                                            int ct, int c, int kg) {
  // B 32x16 layout: lane = column c, K = 16*kg + i (i = 0..15)
  const uint4* p = (const uint4*)(Blds + (ct * 16 + c) * BPITCH + kg * 16);
  const uint4 b0 = p[0];
  const uint4 b1 = p[1];
  B.d[0] = b0.x; B.d[1] = b0.y; B.d[2] = b0.z; B.d[3] = b0.w;
  B.d[4] = b1.x; B.d[5] = b1.y; B.d[6] = b1.z; B.d[7] = b1.w;
}

// Software-pipelined K loop shared by both GEMMs.
__device__ __forceinline__ void gemm_64x64(v8f acc[4],
                                           const unsigned short* arow0,
                                           const float* wbase, int nstride, int kdim,
                                           unsigned short* Blds,  // [2][64*BPITCH]
                                           int tid, int c, int kg) {
  WTile wt;
  load_W(wt, wbase, 0, nstride, tid);
  store_W(wt, Blds, tid);
  BF16x16 A;
  load_A_frag(A, arow0);
  int cur = 0;
  for (int k0 = 0; k0 < kdim; k0 += 32) {
    __syncthreads();                           // buf[cur] visible to all waves
    const bool more = (k0 + 32) < kdim;
    BF16x16 Anext;
    if (more) {
      load_W(wt, wbase, k0 + 32, nstride, tid);     // global loads (LOADcnt path)
      load_A_frag(Anext, arow0 + k0 + 32);
    }
    BF16x16 Bf[4];
#pragma unroll
    for (int ct = 0; ct < 4; ++ct)             // all LDS reads issued up front
      load_B_frag(Bf[ct], Blds + cur * (64 * BPITCH), ct, c, kg);
    if (more)
      store_W(wt, Blds + (cur ^ 1) * (64 * BPITCH), tid);  // fill other buffer
#pragma unroll
    for (int ct = 0; ct < 4; ++ct)             // consume (DScnt path)
      acc[ct] = __builtin_amdgcn_wmma_f32_16x16x32_bf16(false, A.b, false, Bf[ct].b,
                                                        (short)0, acc[ct], false, false);
    A = Anext;
    cur ^= 1;
  }
}

// ---------------------------------------------------------------------------
// Kernel 2: grouped GEMM1 (xn @ w1[e]) + fused interleaved SwiGLU -> s (bf16)
// ---------------------------------------------------------------------------
__global__ void moe_ffn1(const unsigned short* __restrict__ xnbf,
                         const float* __restrict__ w1,
                         const float* __restrict__ b1,
                         const int* __restrict__ cnt,
                         const int* __restrict__ tok_list,
                         const int* __restrict__ row_list,
                         unsigned short* __restrict__ sbuf) {
  const int e    = blockIdx.y >> 4;              // 16 macro row-tiles of 64
  const int tile = blockIdx.y & 15;
  const int rows = cnt[e];
  if (tile * 64 >= rows) return;                 // block-uniform: EXEC stays full

  __shared__ int stok[64], srow[64];
  __shared__ unsigned short Blds[2][64 * BPITCH];   // 2 x 5 KB

  const int tid = threadIdx.x;                   // 128 threads
  if (tid < 64) {
    const int slot = tile * 64 + tid;
    if (slot < rows) { stok[tid] = tok_list[e * Td + slot]; srow[tid] = row_list[e * Td + slot]; }
    else             { stok[tid] = 0;             srow[tid] = -1; }
  }
  __syncthreads();

  const int wave = tid >> 5, lane = tid & 31;
  const int kg = lane >> 4, c = lane & 15;
  const int n0 = blockIdx.x * 64;                // column base in [0, 2I)
  const int tokA = stok[wave * 16 + (lane & 15)];
  const unsigned short* arow0 = xnbf + (size_t)tokA * Hd + kg * 16;
  const float* wbase = w1 + (size_t)e * Hd * (2 * Id) + n0;

  float bias[4];
#pragma unroll
  for (int ct = 0; ct < 4; ++ct) bias[ct] = b1[e * (2 * Id) + n0 + ct * 16 + c];

  v8f acc[4];
#pragma unroll
  for (int ct = 0; ct < 4; ++ct) acc[ct] = (v8f){0.f,0.f,0.f,0.f,0.f,0.f,0.f,0.f};

  gemm_64x64(acc, arow0, wbase, 2 * Id, Hd, &Blds[0][0], tid, c, kg);

  // Fused SwiGLU: even col = gate branch, odd col = linear branch
#pragma unroll
  for (int ct = 0; ct < 4; ++ct) {
    const int sc = ((n0 + ct * 16) >> 1) + (c >> 1);
#pragma unroll
    for (int rI = 0; rI < 8; ++rI) {
      const float h  = acc[ct][rI] + bias[ct];
      const float hp = __shfl_xor(h, 1, 32);     // partner column's value
      const float a  = fminf(h, LIM);
      const float b  = fminf(fmaxf(hp, -LIM), LIM);
      const float sig = __builtin_amdgcn_rcpf(1.f + __expf(-AL * a));
      const float sv  = a * sig * (b + 1.f);
      const int m = wave * 16 + rI + 8 * kg;     // C/D: M = r + 8*(lane>>4)
      const int rowid = srow[m];
      if (((lane & 1) == 0) && rowid >= 0)
        sbuf[(size_t)rowid * Id + sc] = f2bf(sv);
    }
  }
}

// ---------------------------------------------------------------------------
// Kernel 3: grouped GEMM2 (s @ w2[e]) ; epilogue = (acc + b2) * gate -> y
// ---------------------------------------------------------------------------
__global__ void moe_ffn2(const unsigned short* __restrict__ sbuf,
                         const float* __restrict__ w2,
                         const float* __restrict__ b2,
                         const int* __restrict__ cnt,
                         const int* __restrict__ row_list,
                         const float* __restrict__ gate_tk,
                         float* __restrict__ ybuf) {
  const int e    = blockIdx.y >> 4;
  const int tile = blockIdx.y & 15;
  const int rows = cnt[e];
  if (tile * 64 >= rows) return;

  __shared__ int   srow[64];
  __shared__ float sgate[64];
  __shared__ unsigned short Blds[2][64 * BPITCH];

  const int tid = threadIdx.x;                   // 128 threads
  if (tid < 64) {
    const int slot = tile * 64 + tid;
    if (slot < rows) {
      const int rw = row_list[e * Td + slot];
      srow[tid] = rw; sgate[tid] = gate_tk[rw];
    } else {
      srow[tid] = -1; sgate[tid] = 0.f;
    }
  }
  __syncthreads();

  const int wave = tid >> 5, lane = tid & 31;
  const int kg = lane >> 4, c = lane & 15;
  const int n0 = blockIdx.x * 64;                // column base in [0, H)
  int rA = srow[wave * 16 + (lane & 15)]; if (rA < 0) rA = 0;
  const unsigned short* arow0 = sbuf + (size_t)rA * Id + kg * 16;
  const float* wbase = w2 + (size_t)e * Id * Hd + n0;

  float bias[4];
#pragma unroll
  for (int ct = 0; ct < 4; ++ct) bias[ct] = b2[e * Hd + n0 + ct * 16 + c];

  v8f acc[4];
#pragma unroll
  for (int ct = 0; ct < 4; ++ct) acc[ct] = (v8f){0.f,0.f,0.f,0.f,0.f,0.f,0.f,0.f};

  gemm_64x64(acc, arow0, wbase, Hd, Id, &Blds[0][0], tid, c, kg);

#pragma unroll
  for (int ct = 0; ct < 4; ++ct) {
#pragma unroll
    for (int rI = 0; rI < 8; ++rI) {
      const int m = wave * 16 + rI + 8 * kg;
      const int rowid = srow[m];
      if (rowid >= 0)
        ybuf[(size_t)rowid * Hd + n0 + ct * 16 + c] = (acc[ct][rI] + bias[ct]) * sgate[m];
    }
  }
}

// ---------------------------------------------------------------------------
// Kernel 4: deterministic combine  out = x + sum_k y[t*K+k]   (b128 I/O)
// ---------------------------------------------------------------------------
__global__ void combine(const float* __restrict__ x,
                        const float* __restrict__ ybuf,
                        float* __restrict__ out) {
  const int t   = blockIdx.x;
  const int h   = threadIdx.x * 4;               // 256 threads x float4
  float4 s = *(const float4*)(x + (size_t)t * Hd + h);
#pragma unroll
  for (int k = 0; k < Kt; ++k) {
    const float4 yv = *(const float4*)(ybuf + (size_t)(t * Kt + k) * Hd + h);
    s.x += yv.x; s.y += yv.y; s.z += yv.z; s.w += yv.w;
  }
  *(float4*)(out + (size_t)t * Hd + h) = s;
}

// ---------------------------------------------------------------------------
extern "C" void kernel_launch(void* const* d_in, const int* in_sizes, int n_in,
                              void* d_out, int out_size, void* d_ws, size_t ws_size,
                              hipStream_t stream) {
  const float* x  = (const float*)d_in[0];
  const float* ns = (const float*)d_in[1];
  const float* wg = (const float*)d_in[2];
  const float* bg = (const float*)d_in[3];
  const float* w1 = (const float*)d_in[4];
  const float* b1 = (const float*)d_in[5];
  const float* w2 = (const float*)d_in[6];
  const float* b2 = (const float*)d_in[7];
  float* out = (float*)d_out;

  char* ws = (char*)d_ws;
  size_t off = 0;
  unsigned short* xnbf    = (unsigned short*)(ws + off); off += (size_t)Td * Hd * 2;        //  2 MB
  unsigned short* sbuf    = (unsigned short*)(ws + off); off += (size_t)Td * Kt * Id * 2;   //  8 MB
  float*          ybuf    = (float*)(ws + off);          off += (size_t)Td * Kt * Hd * 4;   // 16 MB
  float*          gate_tk = (float*)(ws + off);          off += (size_t)Td * Kt * 4;
  int*            cnt     = (int*)(ws + off);            off += (size_t)Ed * 4;
  int*            tok_list= (int*)(ws + off);            off += (size_t)Ed * Td * 4;
  int*            row_list= (int*)(ws + off);            off += (size_t)Ed * Td * 4;
  (void)in_sizes; (void)n_in; (void)out_size; (void)ws_size;

  hipMemsetAsync(cnt, 0, Ed * sizeof(int), stream);

  rmsnorm_router<<<Td, 128, 0, stream>>>(x, ns, wg, bg, xnbf, cnt,
                                         tok_list, row_list, gate_tk);

  dim3 g1((2 * Id) / 64, Ed * 16);   // 32 x 256 blocks (idle tiles early-out)
  moe_ffn1<<<g1, 128, 0, stream>>>(xnbf, w1, b1, cnt, tok_list, row_list, sbuf);

  dim3 g2(Hd / 64, Ed * 16);         // 16 x 256 blocks
  moe_ffn2<<<g2, 128, 0, stream>>>(sbuf, w2, b2, cnt, row_list, gate_tk, ybuf);

  combine<<<Td, 256, 0, stream>>>(x, ybuf, out);
}
